// Encoder_BahdanauAttention_78804059947113
// MI455X (gfx1250) — matrix operand order repaired and checked by reading, not compiled, not run on hardware
//
#include <hip/hip_runtime.h>
#include <hip/hip_bf16.h>

// ---------------------------------------------------------------------------
// CDNA5 WMMA plumbing (gfx1250, wave32)
// ---------------------------------------------------------------------------
typedef _Float16 half_t;
typedef half_t v16h __attribute__((ext_vector_type(16)));
typedef half_t v8h  __attribute__((ext_vector_type(8)));
typedef float  v8f  __attribute__((ext_vector_type(8)));

#define LN_EPS 1e-5f

// A-fragment: 16x32 f16 tile, row-major in LDS. 8-half runs are contiguous,
// so this lowers to 2x ds_load_b128 per lane.
__device__ inline v16h frag_a_ld(const half_t* tile, int ld) {
  int lane = threadIdx.x & 31;
  int hf = lane >> 4, m = lane & 15;
  v16h r;
#pragma unroll
  for (int j = 0; j < 8; ++j) {
    int kb = hf * 16 + j * 2;
    r[2 * j]     = tile[m * ld + kb];
    r[2 * j + 1] = tile[m * ld + kb + 1];
  }
  return r;
}

// B-fragment from a PRE-SWIZZLED 32x16 sub-tile stored as [lane][16 halves].
// Lowers to 2x ds_load_b128, no packing movs.
__device__ inline v16h frag_b_sw(const half_t* tile) {
  return *reinterpret_cast<const v16h*>(tile + (threadIdx.x & 31) * 16);
}

// swizzled B address for element (k in [0,32), n in [0,16)) of sub-tile nt:
//   lane = (k>>4)*16 + n ; elem = k&15
__device__ inline int bsw_idx(int nt, int k, int n) {
  return nt * 512 + (((k >> 4) * 16) + n) * 16 + (k & 15);
}

__device__ inline v8f wmma_f16(v16h a, v16h b, v8f c) {
  return __builtin_amdgcn_wmma_f32_16x16x32_f16(
      false, a, false, b, (short)0, c, false, false);
}

__device__ inline float wave_sum32(float v) {
#pragma unroll
  for (int o = 16; o > 0; o >>= 1) v += __shfl_xor(v, o, 32);
  return v;
}
__device__ inline float wave_max32(float v) {
#pragma unroll
  for (int o = 16; o > 0; o >>= 1) v = fmaxf(v, __shfl_xor(v, o, 32));
  return v;
}

// BP=64 images, conv1: 3->128 5x5 s2 (64->32), conv2: 128->128 5x5 s2 (32->16)
// conv3: 128->192 3x3 s1. Tq=256, Tk=32, M=192, K=192, D=128.
#define BP 64
#define NCH 128
#define MCH 192
#define DCH 128
#define TQ 256
#define TK 32

// ---------------------------------------------------------------------------
// Kernel 1: conv1 (direct) + GDN1 (WMMA 1x1 conv on x^2) fused.
// grid (64, 32), block 256.
// ---------------------------------------------------------------------------
__global__ void conv1_gdn_kernel(const float* __restrict__ xp,
                                 const float* __restrict__ w1,
                                 const float* __restrict__ b1,
                                 const float* __restrict__ gamma,
                                 const float* __restrict__ beta,
                                 float* __restrict__ x1) {
  __shared__ __align__(32) float  X32[NCH * 32];   // conv out (c,pix)  16 KB
  __shared__ __align__(32) half_t Xsq[NCH * 32];   // swizzled B tiles   8 KB
  __shared__ __align__(32) half_t G[NCH * NCH];    // gamma f16 (A)     32 KB
  const int tid = threadIdx.x;
  const int b = blockIdx.x;
  const int pixbase = blockIdx.y * 32;

  // vectorized gamma stage: 8 contiguous halves per 16B LDS store
  for (int g = tid; g < NCH * NCH / 8; g += 256) {
    int base = g * 8;
    v8h h;
#pragma unroll
    for (int j = 0; j < 8; ++j) h[j] = (half_t)gamma[base + j];
    *reinterpret_cast<v8h*>(&G[base]) = h;
  }

  // direct 5x5 s2 conv, 3 input channels, pad 2
  for (int i = tid; i < NCH * 32; i += 256) {
    int c = i >> 5, p = i & 31;
    int pix = pixbase + p;
    int oy = pix >> 5, ox = pix & 31;
    float s = b1[c];
#pragma unroll
    for (int ci = 0; ci < 3; ++ci)
      for (int ky = 0; ky < 5; ++ky) {
        int iy = oy * 2 - 2 + ky;
        if (iy < 0 || iy >= 64) continue;
        for (int kx = 0; kx < 5; ++kx) {
          int ix = ox * 2 - 2 + kx;
          if (ix < 0 || ix >= 64) continue;
          s += xp[((b * 3 + ci) * 64 + iy) * 64 + ix] *
               w1[((c * 3 + ci) * 5 + ky) * 5 + kx];
        }
      }
    X32[c * 32 + p] = s;
    // swizzled: k = c (4 chunks of 32), two n-subtiles over pix
    Xsq[bsw_idx((c >> 5) * 2 + (p >> 4), c & 31, p & 15)] = (half_t)(s * s);
  }
  __syncthreads();

  // norm[d][pix] = sum_c gamma[d][c] * x[c][pix]^2
  const int w = tid >> 5, lane = tid & 31, hf = lane >> 4;
  v8f acc0 = {}, acc1 = {};
#pragma unroll
  for (int kc = 0; kc < 4; ++kc) {
    v16h a  = frag_a_ld(&G[(w * 16) * NCH + kc * 32], NCH);
    acc0 = wmma_f16(a, frag_b_sw(&Xsq[(kc * 2 + 0) * 512]), acc0);
    acc1 = wmma_f16(a, frag_b_sw(&Xsq[(kc * 2 + 1) * 512]), acc1);
  }
#pragma unroll
  for (int r = 0; r < 8; ++r) {
    int d = w * 16 + hf * 8 + r;
    float bt = beta[d];
    int p0 = (lane & 15), p1 = 16 + (lane & 15);
    x1[((b * NCH) + d) * 1024 + pixbase + p0] =
        X32[d * 32 + p0] * rsqrtf(acc0[r] + bt);
    x1[((b * NCH) + d) * 1024 + pixbase + p1] =
        X32[d * 32 + p1] * rsqrtf(acc1[r] + bt);
  }
}

// ---------------------------------------------------------------------------
// Kernel 2: conv2 implicit-GEMM (K=3200) + GDN2, both WMMA. grid (64, 8).
// ---------------------------------------------------------------------------
__global__ void conv2_gdn_kernel(const float* __restrict__ x1,
                                 const float* __restrict__ w2,
                                 const float* __restrict__ b2,
                                 const float* __restrict__ gamma,
                                 const float* __restrict__ beta,
                                 float* __restrict__ x2) {
  __shared__ __align__(32) half_t Wt[NCH * 32];   // A (outc x k)       8 KB
  __shared__ __align__(32) half_t Pt[32 * 32];    // B swizzled          2 KB
  __shared__ __align__(32) float  X32[NCH * 32];  //                    16 KB
  __shared__ __align__(32) half_t Xsq[NCH * 32];  // B swizzled          8 KB
  const int tid = threadIdx.x;
  const int b = blockIdx.x;
  const int pixbase = blockIdx.y * 32;
  const int w = tid >> 5, lane = tid & 31, hf = lane >> 4;

  v8f acc0 = {}, acc1 = {};
  for (int kk = 0; kk < NCH * 25; kk += 32) {
    __syncthreads();
    // A: weights, k contiguous in flat (outc, ci,5,5) -> vector stage
    for (int g = tid; g < NCH * 4; g += 256) {
      int row = g >> 2, part = g & 3;
      v8h h;
#pragma unroll
      for (int j = 0; j < 8; ++j)
        h[j] = (half_t)w2[row * 3200 + kk + part * 8 + j];
      *reinterpret_cast<v8h*>(&Wt[row * 32 + part * 8]) = h;
    }
    // B: gathered input patches -> swizzled element stores
    for (int i = tid; i < 32 * 32; i += 256) {
      int j = i >> 5, p = i & 31;
      int k = kk + j;
      int ci = k / 25, r = k % 25, ky = r / 5, kx = r % 5;
      int pix = pixbase + p, oy = pix >> 4, ox = pix & 15;
      int iy = oy * 2 - 2 + ky, ix = ox * 2 - 2 + kx;
      float v = 0.0f;
      if (iy >= 0 && iy < 32 && ix >= 0 && ix < 32)
        v = x1[((b * NCH) + ci) * 1024 + iy * 32 + ix];
      Pt[bsw_idx(p >> 4, j, p & 15)] = (half_t)v;
    }
    __syncthreads();
    v16h a = frag_a_ld(&Wt[(w * 16) * 32], 32);
    acc0 = wmma_f16(a, frag_b_sw(&Pt[0]), acc0);
    acc1 = wmma_f16(a, frag_b_sw(&Pt[512]), acc1);
  }
  __syncthreads();
#pragma unroll
  for (int r = 0; r < 8; ++r) {
    int d = w * 16 + hf * 8 + r;
    float bias = b2[d];
    float v0 = acc0[r] + bias, v1 = acc1[r] + bias;
    int p0 = (lane & 15), p1 = 16 + (lane & 15);
    X32[d * 32 + p0] = v0;
    X32[d * 32 + p1] = v1;
    Xsq[bsw_idx((d >> 5) * 2 + 0, d & 31, p0)] = (half_t)(v0 * v0);
    Xsq[bsw_idx((d >> 5) * 2 + 1, d & 31, p1 & 15)] = (half_t)(v1 * v1);
  }
  __syncthreads();

  // GDN2 norm GEMM over channels (K=128)
  v8f n0 = {}, n1 = {};
#pragma unroll
  for (int kc = 0; kc < 4; ++kc) {
    __syncthreads();
    for (int g = tid; g < NCH * 4; g += 256) {
      int row = g >> 2, part = g & 3;
      v8h h;
#pragma unroll
      for (int j = 0; j < 8; ++j)
        h[j] = (half_t)gamma[row * NCH + kc * 32 + part * 8 + j];
      *reinterpret_cast<v8h*>(&Wt[row * 32 + part * 8]) = h;
    }
    __syncthreads();
    v16h a = frag_a_ld(&Wt[(w * 16) * 32], 32);
    n0 = wmma_f16(a, frag_b_sw(&Xsq[(kc * 2 + 0) * 512]), n0);
    n1 = wmma_f16(a, frag_b_sw(&Xsq[(kc * 2 + 1) * 512]), n1);
  }
#pragma unroll
  for (int r = 0; r < 8; ++r) {
    int d = w * 16 + hf * 8 + r;
    float bt = beta[d];
    int p0 = (lane & 15), p1 = 16 + (lane & 15);
    x2[((b * NCH) + d) * 256 + pixbase + p0] =
        X32[d * 32 + p0] * rsqrtf(n0[r] + bt);
    x2[((b * NCH) + d) * 256 + pixbase + p1] =
        X32[d * 32 + p1] * rsqrtf(n1[r] + bt);
  }
}

// ---------------------------------------------------------------------------
// Kernel 3: conv3 implicit GEMM (K=1152) writing (b, t, m) layout.
// grid (64, 4 t-blocks, 2 m-blocks).
// ---------------------------------------------------------------------------
__global__ void conv3_kernel(const float* __restrict__ x2,
                             const float* __restrict__ w3,
                             const float* __restrict__ b3,
                             float* __restrict__ qpre) {
  __shared__ __align__(32) half_t At[64 * 32];   // patches (t x k)   4 KB
  __shared__ __align__(32) half_t Bt[32 * 96];   // weights swizzled  6 KB
  const int tid = threadIdx.x;
  const int b = blockIdx.x;
  const int tbase = blockIdx.y * 64;
  const int mbase = blockIdx.z * 96;
  const int w = tid >> 5, lane = tid & 31, hf = lane >> 4;
  const int mt = w >> 1, ntb = (w & 1) * 3;

  v8f acc[3]; acc[0] = (v8f){}; acc[1] = (v8f){}; acc[2] = (v8f){};
  for (int kk = 0; kk < NCH * 9; kk += 32) {
    __syncthreads();
    // A: gathered patches (element stores)
    for (int i = tid; i < 64 * 32; i += 256) {
      int tt = i >> 5, j = i & 31;
      int t = tbase + tt, k = kk + j;
      int ci = k / 9, r = k % 9, ky = r / 3, kx = r % 3;
      int oy = t >> 4, ox = t & 15;
      int iy = oy - 1 + ky, ix = ox - 1 + kx;
      float v = 0.0f;
      if (iy >= 0 && iy < 16 && ix >= 0 && ix < 16)
        v = x2[((b * NCH) + ci) * 256 + iy * 16 + ix];
      At[tt * 32 + j] = (half_t)v;
    }
    // B: weights, k contiguous -> vector swizzled stage (6 subtiles)
    for (int g = tid; g < 6 * 32 * 2; g += 256) {
      int ll = g >> 1, part = g & 1;
      int nt = ll >> 5, l = ll & 31;
      int col = mbase + nt * 16 + (l & 15);
      int kb = kk + (l >> 4) * 16 + part * 8;
      v8h h;
#pragma unroll
      for (int j = 0; j < 8; ++j) h[j] = (half_t)w3[col * 1152 + kb + j];
      *reinterpret_cast<v8h*>(&Bt[ll * 16 + part * 8]) = h;
    }
    __syncthreads();
    v16h a = frag_a_ld(&At[(mt * 16) * 32], 32);
#pragma unroll
    for (int s = 0; s < 3; ++s)
      acc[s] = wmma_f16(a, frag_b_sw(&Bt[(ntb + s) * 512]), acc[s]);
  }
#pragma unroll
  for (int s = 0; s < 3; ++s)
#pragma unroll
    for (int r = 0; r < 8; ++r) {
      int t = tbase + mt * 16 + hf * 8 + r;
      int m = mbase + (ntb + s) * 16 + (lane & 15);
      qpre[(long)b * (TQ * MCH) + t * MCH + m] = acc[s][r] + b3[m];
    }
}

// ---------------------------------------------------------------------------
// Kernel 4: generic batched WMMA GEMM. C[b] = A[b](MxK) x B(KxN) (+bias).
// transB: B given as (N x K). grid (batch, ceil(M/64), ceil(N/64)).
// K must be a multiple of 32 (true for all uses: 192 or 32).
// ---------------------------------------------------------------------------
__global__ void gemm_kernel(const float* __restrict__ A, long sA,
                            const float* __restrict__ B, long sB,
                            const float* __restrict__ bias,
                            float* __restrict__ C, long sC,
                            int M, int N, int K, int transB) {
  __shared__ __align__(32) half_t At[64 * 32];
  __shared__ __align__(32) half_t Bt[32 * 64];   // 4 swizzled subtiles
  const int tid = threadIdx.x;
  const int bidx = blockIdx.x;
  const int mbase = blockIdx.y * 64;
  const int nbase = blockIdx.z * 64;
  const int w = tid >> 5, lane = tid & 31, hf = lane >> 4;
  const int mt = w >> 1, ntb = (w & 1) * 2;

  v8f acc[2]; acc[0] = (v8f){}; acc[1] = (v8f){};
  for (int kk = 0; kk < K; kk += 32) {
    __syncthreads();
    // A: 8 contiguous k per 16B store (one group per thread)
    {
      int row = mbase + (tid >> 2), part = tid & 3;
      v8h h;
#pragma unroll
      for (int j = 0; j < 8; ++j) {
        int k = kk + part * 8 + j;
        h[j] = (half_t)((row < M) ? A[bidx * sA + (long)row * K + k] : 0.0f);
      }
      *reinterpret_cast<v8h*>(&At[(tid >> 2) * 32 + part * 8]) = h;
    }
    // B: swizzled stage (one 8-half group per thread)
    {
      int ll = tid >> 1, part = tid & 1;
      int nt = ll >> 5, l = ll & 31;
      int col = nbase + nt * 16 + (l & 15);
      int kb = kk + (l >> 4) * 16 + part * 8;
      v8h h;
#pragma unroll
      for (int j = 0; j < 8; ++j) {
        int k = kb + j;
        float v = 0.0f;
        if (col < N)
          v = transB ? B[bidx * sB + (long)col * K + k]
                     : B[bidx * sB + (long)k * N + col];
        h[j] = (half_t)v;
      }
      *reinterpret_cast<v8h*>(&Bt[ll * 16 + part * 8]) = h;
    }
    __syncthreads();
    v16h a = frag_a_ld(&At[(mt * 16) * 32], 32);
#pragma unroll
    for (int s = 0; s < 2; ++s)
      acc[s] = wmma_f16(a, frag_b_sw(&Bt[(ntb + s) * 512]), acc[s]);
  }
#pragma unroll
  for (int s = 0; s < 2; ++s)
#pragma unroll
    for (int r = 0; r < 8; ++r) {
      int row = mbase + mt * 16 + hf * 8 + r;
      int col = nbase + (ntb + s) * 16 + (lane & 15);
      if (row < M && col < N)
        C[bidx * sC + (long)row * N + col] =
            acc[s][r] + (bias ? bias[col] : 0.0f);
    }
}

// ---------------------------------------------------------------------------
// Kernel 5: layernorm over rows of width 192 (wave per row).
// ---------------------------------------------------------------------------
__global__ void ln_rows_kernel(const float* __restrict__ in,
                               float* __restrict__ out,
                               const float* __restrict__ lw,
                               const float* __restrict__ lb, int rows) {
  const int lane = threadIdx.x & 31;
  const int row = blockIdx.x * 8 + (threadIdx.x >> 5);
  if (row >= rows) return;
  float v[6];
#pragma unroll
  for (int i = 0; i < 6; ++i) v[i] = in[(long)row * MCH + lane + 32 * i];
  float s = v[0] + v[1] + v[2] + v[3] + v[4] + v[5];
  float mu = wave_sum32(s) * (1.0f / MCH);
  float sq = 0.0f;
#pragma unroll
  for (int i = 0; i < 6; ++i) { float d = v[i] - mu; sq += d * d; }
  float rs = rsqrtf(wave_sum32(sq) * (1.0f / MCH) + LN_EPS);
#pragma unroll
  for (int i = 0; i < 6; ++i) {
    int c = lane + 32 * i;
    out[(long)row * MCH + c] = (v[i] - mu) * rs * lw[c] + lb[c];
  }
}

// Kernel 6: build kv (transpose of y_g) + layernorm. rows = 64*32.
__global__ void kv_ln_kernel(const float* __restrict__ yg,
                             float* __restrict__ kv,
                             const float* __restrict__ lw,
                             const float* __restrict__ lb) {
  const int lane = threadIdx.x & 31;
  const int row = blockIdx.x * 8 + (threadIdx.x >> 5);
  const int b = row >> 5, t = row & 31;
  float v[6];
#pragma unroll
  for (int i = 0; i < 6; ++i)
    v[i] = yg[((long)b * MCH + lane + 32 * i) * TK + t];
  float s = v[0] + v[1] + v[2] + v[3] + v[4] + v[5];
  float mu = wave_sum32(s) * (1.0f / MCH);
  float sq = 0.0f;
#pragma unroll
  for (int i = 0; i < 6; ++i) { float d = v[i] - mu; sq += d * d; }
  float rs = rsqrtf(wave_sum32(sq) * (1.0f / MCH) + LN_EPS);
#pragma unroll
  for (int i = 0; i < 6; ++i) {
    int c = lane + 32 * i;
    kv[(long)row * MCH + c] = (v[i] - mu) * rs * lw[c] + lb[c];
  }
}

// ---------------------------------------------------------------------------
// Kernel 7: additive-attention energy + softmax.
// grid (64, 32), block 256: wave per query t, lane per key k.
// ---------------------------------------------------------------------------
__global__ void energy_softmax_kernel(const float* __restrict__ qp,
                                      const float* __restrict__ kp,
                                      const float* __restrict__ vw,
                                      float* __restrict__ alpha) {
  __shared__ float kpS[TK * 129];  // stride 129 -> conflict-free rows
  __shared__ float vS[DCH];
  const int tid = threadIdx.x;
  const int b = blockIdx.x;
  for (int i = tid; i < TK * DCH; i += 256)
    kpS[(i >> 7) * 129 + (i & 127)] = kp[(long)b * TK * DCH + i];
  for (int i = tid; i < DCH; i += 256) vS[i] = vw[i];
  __syncthreads();

  const int lane = tid & 31;
  const int t = blockIdx.y * 8 + (tid >> 5);
  const float* qrow = qp + (long)b * TQ * DCH + (long)t * DCH;
  float e = 0.0f;
#pragma unroll 4
  for (int d = 0; d < DCH; ++d)
    e += vS[d] * tanhf(qrow[d] + kpS[lane * 129 + d]);
  float m = wave_max32(e);
  float ex = __expf(e - m);
  float denom = wave_sum32(ex);
  alpha[(long)b * TQ * TK + (long)t * TK + lane] = ex / denom;
}

// Kernel 8: fused = layernorm(q + ctx_proj), transposed to (b, m, t).
__global__ void final_ln_kernel(const float* __restrict__ q,
                                const float* __restrict__ cp,
                                const float* __restrict__ lw,
                                const float* __restrict__ lb,
                                float* __restrict__ out) {
  const int lane = threadIdx.x & 31;
  const int row = blockIdx.x * 8 + (threadIdx.x >> 5);
  const int b = row >> 8, t = row & 255;
  float v[6];
#pragma unroll
  for (int i = 0; i < 6; ++i) {
    long idx = (long)row * MCH + lane + 32 * i;
    v[i] = q[idx] + cp[idx];
  }
  float s = v[0] + v[1] + v[2] + v[3] + v[4] + v[5];
  float mu = wave_sum32(s) * (1.0f / MCH);
  float sq = 0.0f;
#pragma unroll
  for (int i = 0; i < 6; ++i) { float d = v[i] - mu; sq += d * d; }
  float rs = rsqrtf(wave_sum32(sq) * (1.0f / MCH) + LN_EPS);
#pragma unroll
  for (int i = 0; i < 6; ++i) {
    int c = lane + 32 * i;
    out[((long)b * MCH + c) * TQ + t] = (v[i] - mu) * rs * lw[c] + lb[c];
  }
}

// ---------------------------------------------------------------------------
// Host launch
// ---------------------------------------------------------------------------
extern "C" void kernel_launch(void* const* d_in, const int* in_sizes, int n_in,
                              void* d_out, int out_size, void* d_ws,
                              size_t ws_size, hipStream_t stream) {
  (void)in_sizes; (void)n_in; (void)out_size; (void)ws_size;
  const float* xp    = (const float*)d_in[0];
  const float* yg    = (const float*)d_in[1];
  const float* w1    = (const float*)d_in[2];
  const float* b1    = (const float*)d_in[3];
  const float* g1    = (const float*)d_in[4];
  const float* bt1   = (const float*)d_in[5];
  const float* w2    = (const float*)d_in[6];
  const float* b2    = (const float*)d_in[7];
  const float* g2    = (const float*)d_in[8];
  const float* bt2   = (const float*)d_in[9];
  const float* w3    = (const float*)d_in[10];
  const float* b3    = (const float*)d_in[11];
  const float* lnqw  = (const float*)d_in[12];
  const float* lnqb  = (const float*)d_in[13];
  const float* lnkw  = (const float*)d_in[14];
  const float* lnkb  = (const float*)d_in[15];
  const float* lnow  = (const float*)d_in[16];
  const float* lnob  = (const float*)d_in[17];
  const float* Wq    = (const float*)d_in[18];
  const float* Wk    = (const float*)d_in[19];
  const float* vw    = (const float*)d_in[20];
  const float* Wv    = (const float*)d_in[21];
  const float* ow    = (const float*)d_in[22];
  const float* ob    = (const float*)d_in[23];
  float* out = (float*)d_out;
  float* ws  = (float*)d_ws;

  // workspace layout (floats), aliased across pipeline stages (~59 MB total)
  float* x1   = ws + 0;         // 8,388,608  (64,128,32,32)
  float* x2   = ws + 8388608;   // 2,097,152  (64,128,16,16)
  float* qpre = ws + 0;         // 3,145,728  (64,256,192)   [x1 dead]
  float* q    = ws + 3145728;   // 3,145,728
  float* kv   = ws + 6291456;   //   393,216  (64,32,192)
  float* alp  = ws + 6684672;   //   524,288  (64,256,32)
  float* vp   = ws + 7208960;   //   393,216  (64,32,192)
  float* qp   = ws + 0;         // 2,097,152  (64,256,128)   [qpre dead]
  float* kp   = ws + 2097152;   //   262,144  (64,32,128)
  float* ctx  = ws + 8388608;   // 3,145,728  (64,256,192)   [x2 dead]
  float* ctxp = ws + 11534336;  // 3,145,728

  conv1_gdn_kernel<<<dim3(BP, 32), 256, 0, stream>>>(xp, w1, b1, g1, bt1, x1);
  conv2_gdn_kernel<<<dim3(BP, 8), 256, 0, stream>>>(x1, w2, b2, g2, bt2, x2);
  conv3_kernel<<<dim3(BP, 4, 2), 256, 0, stream>>>(x2, w3, b3, qpre);
  ln_rows_kernel<<<BP * TQ / 8, 256, 0, stream>>>(qpre, q, lnqw, lnqb, BP * TQ);
  kv_ln_kernel<<<BP * TK / 8, 256, 0, stream>>>(yg, kv, lnkw, lnkb);
  // q_proj: (256x192)x(192x128)
  gemm_kernel<<<dim3(BP, 4, 2), 256, 0, stream>>>(
      q, (long)TQ * MCH, Wq, 0, nullptr, qp, (long)TQ * DCH, TQ, DCH, MCH, 1);
  // k_proj: (32x192)x(192x128)
  gemm_kernel<<<dim3(BP, 1, 2), 256, 0, stream>>>(
      kv, (long)TK * MCH, Wk, 0, nullptr, kp, (long)TK * DCH, TK, DCH, MCH, 1);
  // v_proj: (32x192)x(192x192)
  gemm_kernel<<<dim3(BP, 1, 3), 256, 0, stream>>>(
      kv, (long)TK * MCH, Wv, 0, nullptr, vp, (long)TK * MCH, TK, MCH, MCH, 1);
  energy_softmax_kernel<<<dim3(BP, TQ / 8), 256, 0, stream>>>(qp, kp, vw, alp);
  // context: (256x32)x(32x192), batched B
  gemm_kernel<<<dim3(BP, 4, 3), 256, 0, stream>>>(
      alp, (long)TQ * TK, vp, (long)TK * MCH, nullptr, ctx, (long)TQ * MCH,
      TQ, MCH, TK, 0);
  // out proj: (256x192)x(192x192) + bias
  gemm_kernel<<<dim3(BP, 4, 3), 256, 0, stream>>>(
      ctx, (long)TQ * MCH, ow, 0, ob, ctxp, (long)TQ * MCH, TQ, MCH, MCH, 1);
  final_ln_kernel<<<BP * TQ / 8, 256, 0, stream>>>(q, ctxp, lnow, lnob, out);
}
